// SkeletonGAT_50620484550940
// MI455X (gfx1250) — matrix operand order repaired and checked by reading, not compiled, not run on hardware
//
#include <hip/hip_runtime.h>
#include <hip/hip_bf16.h>
#include <stdint.h>

#define GN 200000
#define GE 2000000
#define GB 8000
#define ETOT (GE + GN)
#define NEG_SLOPE 0.2f

typedef __bf16 bf16_t;
typedef __bf16 v16bf __attribute__((ext_vector_type(16)));
typedef float  v8f   __attribute__((ext_vector_type(8)));

// ---- float <-> order-preserving uint encoding (for atomicMax on floats) ----
__device__ __forceinline__ unsigned enc_f(float f) {
  unsigned u = __float_as_uint(f);
  return (u & 0x80000000u) ? ~u : (u | 0x80000000u);
}
__device__ __forceinline__ float dec_f(unsigned k) {
  unsigned u = (k & 0x80000000u) ? (k & 0x7FFFFFFFu) : ~k;
  return __uint_as_float(u);
}
#define ENC_NEG_INF 0x007FFFFFu  // enc(-inf)

__device__ __forceinline__ float lrelu(float x) { return x > 0.f ? x : NEG_SLOPE * x; }

// ---------------- weight f32 -> bf16 conversion ----------------
__global__ void k_convw(const float* __restrict__ W2, const float* __restrict__ W3,
                        bf16_t* __restrict__ Wb2, bf16_t* __restrict__ Wb3) {
  int gid = blockIdx.x * blockDim.x + threadIdx.x;
  if (gid >= 8192) return;
  if (gid < 4096) Wb2[gid] = (bf16_t)W2[gid];
  else            Wb3[gid - 4096] = (bf16_t)W3[gid - 4096];
}

// ---------------- layer-1 GEMM: [N,10] x [10,64]^T (K too small for WMMA) ----------------
__global__ void k_gemm10(const float* __restrict__ x, const float* __restrict__ W,
                         float* __restrict__ Hout) {
  int gid = blockIdx.x * blockDim.x + threadIdx.x;
  if (gid >= GN * 64) return;
  int n = gid >> 6, o = gid & 63;
  float s = 0.f;
#pragma unroll
  for (int k = 0; k < 10; ++k) s += x[n * 10 + k] * W[o * 10 + k];
  Hout[gid] = s;
}

// ---------------- WMMA GEMM: Hout[n,o] = sum_k Xbf[n,k] * Wb[o,k], 64x64 weights ----------------
// block = 256 thr = 8 waves; wave -> (rowTile 0..1, colTile 0..3); block covers 32 nodes x 64 ch
__global__ void __launch_bounds__(256) k_gemm_wmma(const bf16_t* __restrict__ X,
                                                   const bf16_t* __restrict__ Wb,
                                                   float* __restrict__ Hout) {
  int lane = threadIdx.x & 31;
  int wv   = threadIdx.x >> 5;
  int rowTile = wv >> 2;
  int colTile = wv & 3;
  int node0 = blockIdx.x * 32 + rowTile * 16;
  int m  = lane & 15;
  int kh = lane >> 4;
  const bf16_t* Arow = X  + (size_t)(node0 + m) * 64;          // A[m, k] = X[node0+m, k]
  const bf16_t* Brow = Wb + (size_t)(colTile * 16 + m) * 64;   // B[k, n] = W[col0+n, k]
  union U { v16bf v; uint4 q[2]; };
  v8f acc = {};
#pragma unroll
  for (int kb = 0; kb < 64; kb += 32) {
    U a, b;
    // ISA 16-bit A 16x32: elems 0..7 -> K = kh*8 + i ; elems 8..15 -> K = 16 + kh*8 + (i-8)
    a.q[0] = *(const uint4*)(Arow + kb + kh * 8);
    a.q[1] = *(const uint4*)(Arow + kb + 16 + kh * 8);
    // B 32x16: elems 0..7 -> K = kh*8 + i ; elems 8..15 -> K = 16 + kh*8 + (i-8), col = lane&15
    b.q[0] = *(const uint4*)(Brow + kb + kh * 8);
    b.q[1] = *(const uint4*)(Brow + kb + 16 + kh * 8);
    acc = __builtin_amdgcn_wmma_f32_16x16x32_bf16(false, a.v, false, b.v,
                                                  (short)0, acc, false, false);
  }
  // C/D 16x16 f32: VGPR r -> M = r + kh*8, N = lane&15
  float* out = Hout + (size_t)node0 * 64 + colTile * 16 + m;
#pragma unroll
  for (int r = 0; r < 8; ++r) out[(size_t)(r + kh * 8) * 64] = acc[r];
}

// ---------------- per-node attention coefficients + init m/z ----------------
__global__ void k_attn(const float* __restrict__ Hf, const float* __restrict__ aS,
                       const float* __restrict__ aD, float* __restrict__ as_,
                       float* __restrict__ ad_, unsigned* __restrict__ mEnc,
                       float* __restrict__ z, int H, int PH) {
  int gid = blockIdx.x * blockDim.x + threadIdx.x;
  if (gid >= GN * H) return;
  int n = gid / H, h = gid % H;
  float s = 0.f, d = 0.f;
  for (int f = 0; f < PH; ++f) {
    float v = Hf[(size_t)n * 64 + h * PH + f];
    s += v * aS[h * PH + f];
    d += v * aD[h * PH + f];
  }
  as_[gid] = s;
  ad_[gid] = d;
  mEnc[gid] = ENC_NEG_INF;
  z[gid] = 0.f;
}

__device__ __forceinline__ void edge_sd(int edge, const int* ei, int& s, int& d) {
  if (edge < GE) { s = ei[edge]; d = ei[GE + edge]; }
  else           { s = d = edge - GE; }
}

// ---------------- segment max over dst (float atomicMax via ordered-uint) ----------------
__global__ void k_edge_max(const int* __restrict__ ei, const float* __restrict__ as_,
                           const float* __restrict__ ad_, unsigned* __restrict__ mEnc, int H) {
  int gid = blockIdx.x * blockDim.x + threadIdx.x;
  if (gid >= ETOT * H) return;
  int edge = gid / H, h = gid % H;
  int s, d; edge_sd(edge, ei, s, d);
  float a = lrelu(as_[(size_t)s * H + h] + ad_[(size_t)d * H + h]);
  atomicMax(&mEnc[(size_t)d * H + h], enc_f(a));
}

// ---------------- segment sum of exp(e - m) ----------------
__global__ void k_edge_z(const int* __restrict__ ei, const float* __restrict__ as_,
                         const float* __restrict__ ad_, const unsigned* __restrict__ mEnc,
                         float* __restrict__ z, int H) {
  int gid = blockIdx.x * blockDim.x + threadIdx.x;
  if (gid >= ETOT * H) return;
  int edge = gid / H, h = gid % H;
  int s, d; edge_sd(edge, ei, s, d);
  float a = lrelu(as_[(size_t)s * H + h] + ad_[(size_t)d * H + h]);
  float m = dec_f(mEnc[(size_t)d * H + h]);
  atomicAdd(&z[(size_t)d * H + h], __expf(a - m));
}

// ---------------- weighted message scatter: agg[dst] += h[src] * alpha ----------------
// block = 256 thr = 4 edges x 64 channels; alpha computed once per (edge,head) in LDS
__global__ void __launch_bounds__(256) k_scatter(const int* __restrict__ ei,
    const float* __restrict__ Hf, const float* __restrict__ as_, const float* __restrict__ ad_,
    const unsigned* __restrict__ mEnc, const float* __restrict__ z,
    float* __restrict__ agg, int H, int phshift) {
  __shared__ float salpha[4 * 4];
  int t  = threadIdx.x;
  int eL = t >> 6;
  int c  = t & 63;
  int edge = blockIdx.x * 4 + eL;
  bool valid = edge < ETOT;
  int s = 0, d = 0;
  if (valid) edge_sd(edge, ei, s, d);
  if (valid && c < H) {
    float a = lrelu(as_[(size_t)s * H + c] + ad_[(size_t)d * H + c]);
    float m = dec_f(mEnc[(size_t)d * H + c]);
    float e = __expf(a - m);
    salpha[eL * 4 + c] = e / (z[(size_t)d * H + c] + 1e-16f);
  }
  __syncthreads();
  if (valid) {
    int h = c >> phshift;
    atomicAdd(&agg[(size_t)d * 64 + c], Hf[(size_t)s * 64 + c] * salpha[eL * 4 + h]);
  }
}

// ---------------- BN stats: per-channel sum & sumsq ----------------
__global__ void __launch_bounds__(256) k_bnstats(const float* __restrict__ xin,
                                                 float* __restrict__ stats, int n) {
  __shared__ float ls[256], lss[256];
  int t = threadIdx.x, c = t & 63, rg = t >> 6;
  float s = 0.f, ss = 0.f;
  for (int r = blockIdx.x * 4 + rg; r < n; r += gridDim.x * 4) {
    float v = xin[(size_t)r * 64 + c];
    s += v; ss += v * v;
  }
  ls[t] = s; lss[t] = ss;
  __syncthreads();
  if (rg == 0) {
    s  = ls[c]  + ls[64 + c]  + ls[128 + c]  + ls[192 + c];
    ss = lss[c] + lss[64 + c] + lss[128 + c] + lss[192 + c];
    atomicAdd(&stats[c], s);
    atomicAdd(&stats[64 + c], ss);
  }
}

__global__ void k_bnfinal(float* stats, int n) {
  int c = threadIdx.x;
  if (c >= 64) return;
  float inv = 1.f / (float)n;
  float mu = stats[c] * inv;
  float var = stats[64 + c] * inv - mu * mu;
  stats[128 + c] = mu;
  stats[192 + c] = rsqrtf(var + 1e-5f);
}

// mode 0: write bf16 (input to next WMMA GEMM); mode 1: write f32
__global__ void k_bnapply(const float* __restrict__ xin, const float* __restrict__ stats,
                          const float* __restrict__ g, const float* __restrict__ be,
                          bf16_t* __restrict__ obf, float* __restrict__ of32, int mode) {
  int gid = blockIdx.x * blockDim.x + threadIdx.x;
  if (gid >= GN * 64) return;
  int c = gid & 63;
  float y = (xin[gid] - stats[128 + c]) * stats[192 + c] * g[c] + be[c];
  float e = y > 0.f ? y : expm1f(y);
  if (mode == 0) obf[gid] = (bf16_t)e;
  else           of32[gid] = e;
}

// ---------------- global mean pool ----------------
__global__ void k_poolacc(const float* __restrict__ hf, const int* __restrict__ batch,
                          float* __restrict__ psum, float* __restrict__ pcnt) {
  int gid = blockIdx.x * blockDim.x + threadIdx.x;
  if (gid >= GN * 64) return;
  int n = gid >> 6, c = gid & 63;
  int b = batch[n];
  atomicAdd(&psum[(size_t)b * 64 + c], hf[gid]);
  if (c == 0) atomicAdd(&pcnt[b], 1.0f);
}

__global__ void k_pooldiv(const float* __restrict__ psum, const float* __restrict__ pcnt,
                          float* __restrict__ out) {
  int gid = blockIdx.x * blockDim.x + threadIdx.x;
  if (gid >= GB * 64) return;
  int b = gid >> 6;
  out[gid] = psum[gid] / fmaxf(pcnt[b], 1.0f);
}

extern "C" void kernel_launch(void* const* d_in, const int* in_sizes, int n_in,
                              void* d_out, int out_size, void* d_ws, size_t ws_size,
                              hipStream_t stream) {
  (void)in_sizes; (void)n_in; (void)out_size; (void)ws_size;
  const float* x   = (const float*)d_in[0];
  const int*   ei  = (const int*)d_in[1];
  const int*   bat = (const int*)d_in[2];
  const float* W1  = (const float*)d_in[3];
  const float* aS1 = (const float*)d_in[4];
  const float* aD1 = (const float*)d_in[5];
  const float* W2  = (const float*)d_in[7];
  const float* aS2 = (const float*)d_in[8];
  const float* aD2 = (const float*)d_in[9];
  const float* W3  = (const float*)d_in[11];
  const float* aS3 = (const float*)d_in[12];
  const float* aD3 = (const float*)d_in[13];
  const float* g1 = (const float*)d_in[15]; const float* be1 = (const float*)d_in[16];
  const float* g2 = (const float*)d_in[17]; const float* be2 = (const float*)d_in[18];
  const float* g3 = (const float*)d_in[19]; const float* be3 = (const float*)d_in[20];
  float* out = (float*)d_out;

  // ---- workspace carve (aligned 256B) ----
  char* w = (char*)d_ws;
  auto carve = [&](size_t bytes) { char* p = w; w += (bytes + 255) & ~(size_t)255; return p; };
  float*    hF   = (float*)carve((size_t)GN * 64 * 4);   // GEMM output (node features)
  float*    aggF = (float*)carve((size_t)GN * 64 * 4);   // aggregation output
  bf16_t*   xbf  = (bf16_t*)carve((size_t)GN * 64 * 2);  // bf16 activations for WMMA GEMM
  float*    asb  = (float*)carve((size_t)GN * 4 * 4);
  float*    adb  = (float*)carve((size_t)GN * 4 * 4);
  unsigned* mEnc = (unsigned*)carve((size_t)GN * 4 * 4);
  float*    zb   = (float*)carve((size_t)GN * 4 * 4);
  bf16_t*   Wbf2 = (bf16_t*)carve(4096 * 2);
  bf16_t*   Wbf3 = (bf16_t*)carve(4096 * 2);
  float*    stats= (float*)carve(256 * 4);
  float*    psum = (float*)carve((size_t)GB * 64 * 4);
  float*    pcnt = (float*)carve((size_t)GB * 4);

  k_convw<<<(8192 + 255) / 256, 256, 0, stream>>>(W2, W3, Wbf2, Wbf3);

  const float* aSrc[3] = {aS1, aS2, aS3};
  const float* aDst[3] = {aD1, aD2, aD3};
  const float* gArr[3] = {g1, g2, g3};
  const float* bArr[3] = {be1, be2, be3};
  const int    Hh[3]   = {4, 4, 1};
  const int    PHh[3]  = {16, 16, 64};
  const int    phs[3]  = {4, 4, 6};

  for (int L = 0; L < 3; ++L) {
    hipMemsetAsync(aggF, 0, (size_t)GN * 64 * 4, stream);
    hipMemsetAsync(stats, 0, 256 * 4, stream);
    if (L == 0)
      k_gemm10<<<(GN * 64 + 255) / 256, 256, 0, stream>>>(x, W1, hF);
    else
      k_gemm_wmma<<<GN / 32, 256, 0, stream>>>(xbf, L == 1 ? Wbf2 : Wbf3, hF);
    int H = Hh[L], PH = PHh[L];
    int nh = GN * H;
    k_attn<<<(nh + 255) / 256, 256, 0, stream>>>(hF, aSrc[L], aDst[L], asb, adb, mEnc, zb, H, PH);
    int eth = ETOT * H;
    k_edge_max<<<(eth + 255) / 256, 256, 0, stream>>>(ei, asb, adb, mEnc, H);
    k_edge_z<<<(eth + 255) / 256, 256, 0, stream>>>(ei, asb, adb, mEnc, zb, H);
    k_scatter<<<(ETOT + 3) / 4, 256, 0, stream>>>(ei, hF, asb, adb, mEnc, zb, aggF, H, phs[L]);
    k_bnstats<<<512, 256, 0, stream>>>(aggF, stats, GN);
    k_bnfinal<<<1, 64, 0, stream>>>(stats, GN);
    // note: +bias before BN cancels in mean subtraction -> omitted
    k_bnapply<<<(GN * 64 + 255) / 256, 256, 0, stream>>>(aggF, stats, gArr[L], bArr[L],
                                                         xbf, hF, L == 2 ? 1 : 0);
  }

  hipMemsetAsync(psum, 0, (size_t)GB * 64 * 4, stream);
  hipMemsetAsync(pcnt, 0, (size_t)GB * 4, stream);
  k_poolacc<<<(GN * 64 + 255) / 256, 256, 0, stream>>>(hF, bat, psum, pcnt);
  k_pooldiv<<<(GB * 64 + 255) / 256, 256, 0, stream>>>(psum, pcnt, out);
}
